// MixAttention_49744311222622
// MI455X (gfx1250) — matrix operand order repaired and checked by reading, not compiled, not run on hardware
//
#include <hip/hip_runtime.h>
#include <hip/hip_bf16.h>
#include <math.h>

typedef __bf16 bf16;
typedef __attribute__((ext_vector_type(4)))  __bf16 v4bf;
typedef __attribute__((ext_vector_type(16))) __bf16 v16bf;
typedef __attribute__((ext_vector_type(8)))  float  v8f;
typedef __attribute__((ext_vector_type(4)))  unsigned int v4u;
typedef __attribute__((ext_vector_type(8)))  int v8i;
typedef __attribute__((ext_vector_type(4)))  int v4i;

#define B_   4
#define S_   2048
#define D_   512
#define H_   8
#define DH_  64
#define ROWS_ (B_*S_)   // 8192
#define EPS_ 1e-5f

#if defined(__has_builtin)
#if __has_builtin(__builtin_amdgcn_tensor_load_to_lds) && __has_builtin(__builtin_amdgcn_s_wait_tensorcnt)
#define USE_TDM 1
#endif
#endif
#ifndef USE_TDM
#define USE_TDM 0
#endif

static __device__ __forceinline__ v8f wmma_bf16(v16bf a, v16bf b, v8f c) {
  // D = A(16x32 bf16) x B(32x16 bf16) + C(16x16 f32)
  return __builtin_amdgcn_wmma_f32_16x16x32_bf16(false, a, false, b, (short)0, c, false, false);
}

// A fragment (16x32): lane(l,hi) row = l; element e: K = hi*16+e
static __device__ __forceinline__ v16bf load_a_frag(const bf16* p, int hi) {
  v16bf a;
  const bf16* q = p + hi * 16;
#pragma unroll
  for (int e = 0; e < 16; ++e) a[e] = q[e];
  return a;
}
// B fragment (32x16): lane(l,hi) col = l; element e: K = hi*16+e (p = K-major run of this column)
static __device__ __forceinline__ v16bf load_b_frag(const bf16* p, int hi) {
  v16bf b;
  const bf16* q = p + hi * 16;
#pragma unroll
  for (int e = 0; e < 16; ++e) b[e] = q[e];
  return b;
}

static __device__ __forceinline__ v4bf pack4(float4 v) {
  v4bf r;
  r[0] = (bf16)v.x; r[1] = (bf16)v.y; r[2] = (bf16)v.z; r[3] = (bf16)v.w;
  return r;
}

static __device__ __forceinline__ void fetch_tiles_f32(const float* __restrict__ X,
                                                       const float* __restrict__ W,
                                                       int rowBase, int colBase, int kb,
                                                       int tid, float4* av, float4* bv) {
#pragma unroll
  for (int t = 0; t < 4; ++t) {
    int idx = t * 512 + tid * 4;
    int r = idx >> 5, kk = idx & 31;               // A: kk fastest, 4-contig
    av[t] = *(const float4*)&X[(size_t)(rowBase + r) * 512 + kb + kk];
    int kk2 = idx >> 6, c2 = idx & 63;             // B: col fastest, 4-contig
    bv[t] = *(const float4*)&W[(size_t)(kb + kk2) * 512 + colBase + c2];
  }
}

static __device__ __forceinline__ void store_tiles(bf16* As, bf16* Bs, int tid,
                                                   const float4* av, const float4* bv) {
#pragma unroll
  for (int t = 0; t < 4; ++t) {
    int idx = t * 512 + tid * 4;
    *(v4bf*)&As[idx] = pack4(av[t]);               // ds_store_b64
    int kk2 = idx >> 6, c2 = idx & 63;
    Bs[(c2 + 0) * 32 + kk2] = (bf16)bv[t].x;       // transposed scatter for contig B-frags
    Bs[(c2 + 1) * 32 + kk2] = (bf16)bv[t].y;
    Bs[(c2 + 2) * 32 + kk2] = (bf16)bv[t].z;
    Bs[(c2 + 3) * 32 + kk2] = (bf16)bv[t].w;
  }
}

static __device__ __forceinline__ void tile_mma(const bf16* As, const bf16* Bs,
                                                int wm, int wn, int l, int hi, v8f c[2][2]) {
  v16bf a0 = load_a_frag(&As[(wm * 32 + 0  + l) * 32], hi);
  v16bf a1 = load_a_frag(&As[(wm * 32 + 16 + l) * 32], hi);
  v16bf b0 = load_b_frag(&Bs[(wn * 32 + 0  + l) * 32], hi);
  v16bf b1 = load_b_frag(&Bs[(wn * 32 + 16 + l) * 32], hi);
  c[0][0] = wmma_bf16(a0, b0, c[0][0]);
  c[0][1] = wmma_bf16(a0, b1, c[0][1]);
  c[1][0] = wmma_bf16(a1, b0, c[1][0]);
  c[1][1] = wmma_bf16(a1, b1, c[1][1]);
}

#if USE_TDM
// TDM: DMA a 64-row x 32-col bf16 tile (row stride 512 elems) from global into
// LDS, packed row-major (row stride 32 elems) -- exactly the As layout.
// D# per CDNA5 ISA 8.3/8.4: group0 {count|lds|gaddr|type=2}, group1 {dsz=2B,
// tensor 512x8192, tile 32x64, dim0_stride=512}, groups 2/3 zero (2D).
static __device__ __forceinline__ void tdm_load_tile_bf16(const bf16* gsrc, bf16* lds_dst) {
  unsigned long long ga = (unsigned long long)(uintptr_t)gsrc;
  v4u g0;
  g0[0] = 1u;                                          // count=1, user desc
  g0[1] = (unsigned)(uintptr_t)lds_dst;                // lds_addr (low 32 bits = LDS offset)
  g0[2] = (unsigned)(ga & 0xffffffffu);                // global_addr[31:0]
  g0[3] = (unsigned)((ga >> 32) & 0x1ffffffu) | (2u << 30);  // global_addr[56:32] | type=2
  v8i g1;
  g1[0] = 0x10000;        // data_size=1 (2 bytes); mask/flags/pad = 0
  g1[1] = 0x02000000;     // tensor_dim0 = 512  (bits 79:48 -> dw1[31:16])
  g1[2] = 0x20000000;     // tensor_dim1 = 8192 (bits 111:80 -> dw2[31:16])
  g1[3] = 32 << 16;       // tile_dim0 = 32     (bits 127:112)
  g1[4] = 64;             // tile_dim1 = 64     (bits 143:128); tile_dim2 = 0
  g1[5] = 512;            // tensor_dim0_stride = 512 (bits 207:160)
  g1[6] = 0;
  g1[7] = 0;              // tensor_dim1_stride = 0 (unused, 2D)
  v4i z4 = {0, 0, 0, 0};
  v8i z8 = {0, 0, 0, 0, 0, 0, 0, 0};
  // 6-arg form (clang-23 / therock headers): groups 0..3, spare group, cpol
  __builtin_amdgcn_tensor_load_to_lds(g0, g1, z4, z4, z8, 0);
}
#endif

// ---------------------------------------------------------------------------
// Kernel 1: fused projections.
//   p=0..3 : Y = X @ W + b  (K=512) -> Qcat/Kcat [b*H+h][s][off+d] (off 0 / 64)
//   p=4    : Vsum = [Vd|Vt] @ [[Wvd];[Wvt]] + (bvd+bvt)  (K=1024) -> V^T [b*H+h][d][s]
// ---------------------------------------------------------------------------
__global__ __launch_bounds__(128)
void proj_kernel(const float* __restrict__ Qd, const float* __restrict__ Qt,
                 const float* __restrict__ Kd, const float* __restrict__ Kt,
                 const float* __restrict__ Vd, const float* __restrict__ Vt,
                 const float* __restrict__ Wqd, const float* __restrict__ bqd,
                 const float* __restrict__ Wqt, const float* __restrict__ bqt,
                 const float* __restrict__ Wkd, const float* __restrict__ bkd,
                 const float* __restrict__ Wkt, const float* __restrict__ bkt,
                 const float* __restrict__ Wvd, const float* __restrict__ bvd,
                 const float* __restrict__ Wvt, const float* __restrict__ bvt,
                 bf16* __restrict__ Qcat, bf16* __restrict__ Kcat, bf16* __restrict__ VsumT)
{
  const int p = blockIdx.z;
  const float *X0, *X1 = nullptr, *W0, *W1 = nullptr, *bias0, *bias1 = nullptr;
  bf16* dstQK = nullptr; int off = 0;
  switch (p) {
    case 0: X0 = Qd; W0 = Wqd; bias0 = bqd; dstQK = Qcat; off = 0;  break;
    case 1: X0 = Qt; W0 = Wqt; bias0 = bqt; dstQK = Qcat; off = 64; break;
    case 2: X0 = Kd; W0 = Wkd; bias0 = bkd; dstQK = Kcat; off = 0;  break;
    case 3: X0 = Kt; W0 = Wkt; bias0 = bkt; dstQK = Kcat; off = 64; break;
    default: X0 = Vd; X1 = Vt; W0 = Wvd; W1 = Wvt; bias0 = bvd; bias1 = bvt; break;
  }
  const int NK = (p == 4) ? 32 : 16;   // 32-wide k-steps

  __shared__ bf16 As[64 * 32];   // [row][kk]
  __shared__ bf16 Bs[64 * 32];   // [col][kk]

  const int tid = threadIdx.x;
  const int lane = tid & 31;
  const int w = tid >> 5;
  const int l = lane & 15, hi = lane >> 4;
  const int wm = w >> 1, wn = w & 1;
  const int rowBase = blockIdx.y * 64;
  const int colBase = blockIdx.x * 64;

  v8f c[2][2] = {};
  float4 av[4], bv[4];

  fetch_tiles_f32(X0, W0, rowBase, colBase, 0, tid, av, bv);
  store_tiles(As, Bs, tid, av, bv);

  for (int it = 1; it < NK; ++it) {
    const int k0 = it * 32;
    const float* X = (k0 < 512) ? X0 : X1;     // uniform per step; tile never straddles
    const float* W = (k0 < 512) ? W0 : W1;
    fetch_tiles_f32(X, W, rowBase, colBase, k0 & 511, tid, av, bv);  // loads in flight
    __syncthreads();
    tile_mma(As, Bs, wm, wn, l, hi, c);
    __syncthreads();
    store_tiles(As, Bs, tid, av, bv);
  }
  __syncthreads();
  tile_mma(As, Bs, wm, wn, l, hi, c);

#pragma unroll
  for (int i = 0; i < 2; ++i) {
#pragma unroll
    for (int j = 0; j < 2; ++j) {
#pragma unroll
      for (int r = 0; r < 8; ++r) {
        int grow = rowBase + wm * 32 + i * 16 + hi * 8 + r;
        int gcol = colBase + wn * 32 + j * 16 + l;
        float v = c[i][j][r] + bias0[gcol] + (bias1 ? bias1[gcol] : 0.f);
        int bb = grow >> 11;            // / S_
        int s  = grow & (S_ - 1);
        int h  = gcol >> 6, d = gcol & 63;
        if (p < 4) {
          dstQK[(((size_t)(bb * H_ + h) * S_ + s) << 7) + off + d] = (bf16)v;
        } else {
          VsumT[((size_t)(bb * H_ + h) * 64 + d) * S_ + s] = (bf16)v;
        }
      }
    }
  }
}

// ---------------------------------------------------------------------------
// Kernel 2: flash attention (per (b,h); online softmax; 12 WMMA per 16x32 tile)
// ---------------------------------------------------------------------------
__global__ __launch_bounds__(128)
void attn_kernel(const bf16* __restrict__ Qcat, const bf16* __restrict__ Kcat,
                 const bf16* __restrict__ VsumT, bf16* __restrict__ CTX)
{
  const int bh = blockIdx.y;          // b*H + h
  const int tid = threadIdx.x;
  const int w = tid >> 5;
  const int lane = tid & 31;
  const int l = lane & 15, hi = lane >> 4;

  __shared__ bf16 Ps[4][16 * 32];     // per-wave P staging (C-layout -> A-layout)

  const int qbase = blockIdx.x * 64 + w * 16;
  const bf16* Qrow = Qcat + ((size_t)bh * S_ + qbase + l) * 128;
  v16bf qf[4];
#pragma unroll
  for (int kt = 0; kt < 4; ++kt) qf[kt] = load_a_frag(Qrow + kt * 32, hi);

  v8f acc[4] = {};
  float m[8], lsum[8];
#pragma unroll
  for (int r = 0; r < 8; ++r) { m[r] = -3.0e38f; lsum[r] = 0.f; }

  const bf16* Kbase = Kcat + (size_t)bh * S_ * 128;
  const bf16* Vbase = VsumT + (size_t)bh * 64 * S_;

  for (int kb = 0; kb < S_; kb += 32) {
    if (kb + 32 < S_) {  // prefetch next key/value block (-> global_prefetch_b8)
      __builtin_prefetch(Kbase + (size_t)(kb + 32 + lane) * 128, 0, 0);
      __builtin_prefetch(Kbase + (size_t)(kb + 32 + lane) * 128 + 64, 0, 0);
      __builtin_prefetch(Vbase + (size_t)lane * S_ + kb + 32, 0, 0);
      __builtin_prefetch(Vbase + (size_t)(32 + lane) * S_ + kb + 32, 0, 0);
    }
    v8f sc0 = {}, sc1 = {};
#pragma unroll
    for (int kt = 0; kt < 4; ++kt) {
      v16bf b0 = load_b_frag(Kbase + (size_t)(kb + l) * 128 + kt * 32, hi);
      v16bf b1 = load_b_frag(Kbase + (size_t)(kb + 16 + l) * 128 + kt * 32, hi);
      sc0 = wmma_bf16(qf[kt], b0, sc0);
      sc1 = wmma_bf16(qf[kt], b1, sc1);
    }
    const float scale = 0.125f;       // 1/sqrt(64)
    float corr[8];
#pragma unroll
    for (int r = 0; r < 8; ++r) {
      sc0[r] *= scale; sc1[r] *= scale;
      float v = fmaxf(sc0[r], sc1[r]);
#pragma unroll
      for (int mk = 8; mk >= 1; mk >>= 1) v = fmaxf(v, __shfl_xor(v, mk));
      float mn = fmaxf(m[r], v);
      corr[r] = __expf(m[r] - mn);
      m[r] = mn;
      float p0 = __expf(sc0[r] - mn);
      float p1 = __expf(sc1[r] - mn);
      sc0[r] = p0; sc1[r] = p1;
      float ps = p0 + p1;
#pragma unroll
      for (int mk = 8; mk >= 1; mk >>= 1) ps += __shfl_xor(ps, mk);
      lsum[r] = lsum[r] * corr[r] + ps;
    }
#pragma unroll
    for (int nt = 0; nt < 4; ++nt)
#pragma unroll
      for (int r = 0; r < 8; ++r) acc[nt][r] *= corr[r];

#pragma unroll
    for (int r = 0; r < 8; ++r) {
      Ps[w][(r + hi * 8) * 32 + l]      = (bf16)sc0[r];
      Ps[w][(r + hi * 8) * 32 + 16 + l] = (bf16)sc1[r];
    }
    __syncthreads();
    v16bf pa = load_a_frag(&Ps[w][l * 32], hi);
#pragma unroll
    for (int nt = 0; nt < 4; ++nt) {
      v16bf bvv = load_b_frag(Vbase + (size_t)(nt * 16 + l) * S_ + kb, hi);
      acc[nt] = wmma_bf16(pa, bvv, acc[nt]);
    }
    __syncthreads();
  }

  const int b = bh >> 3, h = bh & 7;
#pragma unroll
  for (int nt = 0; nt < 4; ++nt) {
#pragma unroll
    for (int r = 0; r < 8; ++r) {
      int s = qbase + hi * 8 + r;
      float v = acc[nt][r] / lsum[r];
      CTX[((size_t)b * S_ + s) * 512 + h * 64 + nt * 16 + l] = (bf16)v;
    }
  }
}

// ---------------------------------------------------------------------------
// Kernel 3: output projection  preLN = CTX(bf16) @ Wo + bo  (fp32 out)
// A-tiles DMA'd by the Tensor Data Mover into double-buffered LDS (wave 0
// issues, TENSORcnt drains, barrier releases); B-tiles register-pipelined.
// ---------------------------------------------------------------------------
__global__ __launch_bounds__(128)
void out_gemm_kernel(const bf16* __restrict__ CTX, const float* __restrict__ Wo,
                     const float* __restrict__ bo, float* __restrict__ preLN)
{
  __shared__ bf16 As2[2][64 * 32];
  __shared__ bf16 Bs2[2][64 * 32];
  const int tid = threadIdx.x;
  const int lane = tid & 31;
  const int w = tid >> 5;
  const int l = lane & 15, hi = lane >> 4;
  const int wm = w >> 1, wn = w & 1;
  const int rowBase = blockIdx.y * 64;
  const int colBase = blockIdx.x * 64;

  v8f c[2][2] = {};
  float4 bv[4];

  const bf16* Abase = CTX + (size_t)rowBase * 512;

  auto fetchB = [&](int kb) {
#pragma unroll
    for (int t = 0; t < 4; ++t) {
      int idx = t * 512 + tid * 4;
      int kk2 = idx >> 6, c2 = idx & 63;
      bv[t] = *(const float4*)&Wo[(size_t)(kb + kk2) * 512 + colBase + c2];
    }
  };
  auto stageB = [&](bf16* Bs) {
#pragma unroll
    for (int t = 0; t < 4; ++t) {
      int idx = t * 512 + tid * 4;
      int kk2 = idx >> 6, c2 = idx & 63;
      Bs[(c2 + 0) * 32 + kk2] = (bf16)bv[t].x;
      Bs[(c2 + 1) * 32 + kk2] = (bf16)bv[t].y;
      Bs[(c2 + 2) * 32 + kk2] = (bf16)bv[t].z;
      Bs[(c2 + 3) * 32 + kk2] = (bf16)bv[t].w;
    }
  };

#if USE_TDM
  const bool issuer = (w == 0);          // wave-uniform: only wave 0 issues DMA
  if (issuer) tdm_load_tile_bf16(Abase, &As2[0][0]);
  fetchB(0);
  stageB(Bs2[0]);
  if (issuer) __builtin_amdgcn_s_wait_tensorcnt(0);
  __syncthreads();
  for (int it = 1; it < 16; ++it) {
    const int cur = (it - 1) & 1, nxt = it & 1;
    if (issuer) tdm_load_tile_bf16(Abase + it * 32, &As2[nxt][0]);  // DMA overlaps MMA
    fetchB(it * 32);                                                // loads in flight
    tile_mma(As2[cur], Bs2[cur], wm, wn, l, hi, c);
    stageB(Bs2[nxt]);
    if (issuer) __builtin_amdgcn_s_wait_tensorcnt(0);
    __syncthreads();
  }
  tile_mma(As2[1], Bs2[1], wm, wn, l, hi, c);
#else
  // fallback: register-pipelined A staging
  v4bf av[4];
  auto fetchA = [&](int kb) {
#pragma unroll
    for (int t = 0; t < 4; ++t) {
      int idx = t * 512 + tid * 4;
      int r = idx >> 5, kk = idx & 31;
      av[t] = *(const v4bf*)&Abase[(size_t)r * 512 + kb + kk];
    }
  };
  auto stageA = [&](bf16* As) {
#pragma unroll
    for (int t = 0; t < 4; ++t) *(v4bf*)&As[t * 512 + tid * 4] = av[t];
  };
  fetchA(0); fetchB(0);
  stageA(As2[0]); stageB(Bs2[0]);
  __syncthreads();
  for (int it = 1; it < 16; ++it) {
    const int cur = (it - 1) & 1, nxt = it & 1;
    fetchA(it * 32); fetchB(it * 32);
    tile_mma(As2[cur], Bs2[cur], wm, wn, l, hi, c);
    stageA(As2[nxt]); stageB(Bs2[nxt]);
    __syncthreads();
  }
  tile_mma(As2[1], Bs2[1], wm, wn, l, hi, c);
#endif

#pragma unroll
  for (int i = 0; i < 2; ++i)
#pragma unroll
    for (int j = 0; j < 2; ++j)
#pragma unroll
      for (int r = 0; r < 8; ++r) {
        int grow = rowBase + wm * 32 + i * 16 + hi * 8 + r;
        int gcol = colBase + wn * 32 + j * 16 + l;
        preLN[(size_t)grow * 512 + gcol] = c[i][j][r] + bo[gcol];
      }
}

// ---------------------------------------------------------------------------
// Kernel 4: residual add + LayerNorm (one 128-thread block per row of 512)
// ---------------------------------------------------------------------------
__global__ __launch_bounds__(128)
void ln_kernel(const float* __restrict__ preLN, const float* __restrict__ resid,
               const float* __restrict__ gamma, const float* __restrict__ beta,
               float* __restrict__ out)
{
  const size_t row = blockIdx.x;
  const int tid = threadIdx.x;
  float x[4], s = 0.f, s2 = 0.f;
#pragma unroll
  for (int i = 0; i < 4; ++i) {
    int cidx = tid + i * 128;
    float v = preLN[row * 512 + cidx] + resid[row * 512 + cidx];
    x[i] = v; s += v; s2 += v * v;
  }
#pragma unroll
  for (int mk = 16; mk >= 1; mk >>= 1) {
    s  += __shfl_xor(s, mk);
    s2 += __shfl_xor(s2, mk);
  }
  __shared__ float ps[4], ps2[4];
  if ((tid & 31) == 0) { ps[tid >> 5] = s; ps2[tid >> 5] = s2; }
  __syncthreads();
  s  = ps[0] + ps[1] + ps[2] + ps[3];
  s2 = ps2[0] + ps2[1] + ps2[2] + ps2[3];
  const float mu = s * (1.f / 512.f);
  const float var = s2 * (1.f / 512.f) - mu * mu;
  const float rstd = rsqrtf(var + EPS_);
#pragma unroll
  for (int i = 0; i < 4; ++i) {
    int cidx = tid + i * 128;
    out[row * 512 + cidx] = (x[i] - mu) * rstd * gamma[cidx] + beta[cidx];
  }
}

// ---------------------------------------------------------------------------
extern "C" void kernel_launch(void* const* d_in, const int* in_sizes, int n_in,
                              void* d_out, int out_size, void* d_ws, size_t ws_size,
                              hipStream_t stream) {
  (void)in_sizes; (void)n_in; (void)out_size; (void)ws_size;
  const float* Qd  = (const float*)d_in[0];
  const float* Qt  = (const float*)d_in[1];
  const float* Kd  = (const float*)d_in[2];
  const float* Kt  = (const float*)d_in[3];
  const float* Vd  = (const float*)d_in[4];
  const float* Vt  = (const float*)d_in[5];
  const float* Wqd = (const float*)d_in[6];   const float* bqd = (const float*)d_in[7];
  const float* Wqt = (const float*)d_in[8];   const float* bqt = (const float*)d_in[9];
  const float* Wkd = (const float*)d_in[10];  const float* bkd = (const float*)d_in[11];
  const float* Wkt = (const float*)d_in[12];  const float* bkt = (const float*)d_in[13];
  const float* Wvd = (const float*)d_in[14];  const float* bvd = (const float*)d_in[15];
  const float* Wvt = (const float*)d_in[16];  const float* bvt = (const float*)d_in[17];
  const float* Wo  = (const float*)d_in[18];  const float* bo  = (const float*)d_in[19];
  const float* gamma = (const float*)d_in[20];
  const float* beta  = (const float*)d_in[21];
  float* out = (float*)d_out;

  char* ws = (char*)d_ws;
  bf16*  Qcat  = (bf16*)(ws + 0);                      // 16.78 MB
  bf16*  Kcat  = (bf16*)(ws + (size_t)16777216);       // 16.78 MB
  bf16*  VsumT = (bf16*)(ws + (size_t)33554432);       //  8.39 MB
  bf16*  CTX   = (bf16*)(ws + (size_t)41943040);       //  8.39 MB
  float* preLN = (float*)(ws + (size_t)50331648);      // 16.78 MB  (total 64 MB)

  proj_kernel<<<dim3(8, 128, 5), 128, 0, stream>>>(
      Qd, Qt, Kd, Kt, Vd, Vt,
      Wqd, bqd, Wqt, bqt, Wkd, bkd, Wkt, bkt, Wvd, bvd, Wvt, bvt,
      Qcat, Kcat, VsumT);
  attn_kernel<<<dim3(S_ / 64, B_ * H_), 128, 0, stream>>>(Qcat, Kcat, VsumT, CTX);
  out_gemm_kernel<<<dim3(8, 128), 128, 0, stream>>>(CTX, Wo, bo, preLN);
  ln_kernel<<<dim3(ROWS_), 128, 0, stream>>>(preLN, Qd, gamma, beta, out);
}
